// GraphAttentionLayer_26852135534911
// MI455X (gfx1250) — matrix operand order repaired and checked by reading, not compiled, not run on hardware
//
#include <hip/hip_runtime.h>
#include <hip/hip_bf16.h>

typedef __attribute__((ext_vector_type(16))) _Float16 v16h;
typedef __attribute__((ext_vector_type(8)))  _Float16 v8h;
typedef __attribute__((ext_vector_type(8)))  float    v8f;

#define GAT_N      4096
#define GAT_FIN    128
#define GAT_HEADS  4
#define GAT_HID    16
#define GAT_FOUT   64          // HEADS * HID
#define GAT_SLOPE  0.2f
#define GAT_KCHUNK 4           // split-K chunks over j
#define GAT_JLEN   (GAT_N / GAT_KCHUNK)
#define GAT_LOG2E  1.44269504088896340736f

// ---------------------------------------------------------------------------
// Kernel 1a: convert x (f32, row-major 4096x128) -> f16
__global__ void gat_conv_x(const float* __restrict__ x, _Float16* __restrict__ xh) {
    int i = blockIdx.x * 256 + threadIdx.x;          // 4096*128 elements
    xh[i] = (_Float16)x[i];
}

// Kernel 1b: W (f32, 128x64 row-major) -> Wt f16 (64x128 = W^T)
__global__ void gat_conv_w(const float* __restrict__ W, _Float16* __restrict__ wt) {
    int i = blockIdx.x * 256 + threadIdx.x;          // 64*128 elements, i = c*128 + k
    int c = i >> 7, k = i & 127;
    wt[i] = (_Float16)W[k * GAT_FOUT + c];
}

// ---------------------------------------------------------------------------
// Kernel 2: H = x @ W via v_wmma_f32_16x16x32_f16; write transposed Ht (64 x 4096) f16.
__global__ void gat_proj(const _Float16* __restrict__ xh, const _Float16* __restrict__ wt,
                         _Float16* __restrict__ ht) {
    const int wave = threadIdx.x >> 5;
    const int lane = threadIdx.x & 31;
    const int i0   = blockIdx.x * 16;
    const int c0   = wave * 16;
    const int r    = lane & 15;
    const bool hi  = lane >= 16;
    const int row  = i0 + r;                 // A-matrix row for this lane
    const int col  = c0 + r;                 // B/C column for this lane
    const int abase = hi ? 16 : 0;           // A: K chunk base within 32-block
    const int bbase = hi ? 16 : 0;           // B: K chunk base within 32-block

    v8f acc = {};
    #pragma unroll
    for (int kb = 0; kb < GAT_FIN; kb += 32) {
        const _Float16* ap = xh + row * GAT_FIN + kb + abase;
        v8h a0 = *(const v8h*)(ap);          // K = kb+abase    .. +7
        v8h a1 = *(const v8h*)(ap + 8);      // K = kb+abase+8 .. +15
        v16h a;
        #pragma unroll
        for (int e = 0; e < 8; ++e) { a[e] = a0[e]; a[8 + e] = a1[e]; }
        v16h b = *(const v16h*)(wt + col * GAT_FIN + kb + bbase);
        acc = __builtin_amdgcn_wmma_f32_16x16x32_f16(false, a, false, b,
                                                     (short)0, acc, false, false);
    }
    const int orow = i0 + (hi ? 8 : 0);
    v8h o;
    #pragma unroll
    for (int rr = 0; rr < 8; ++rr) o[rr] = (_Float16)acc[rr];
    *(v8h*)(ht + col * GAT_N + orow) = o;
}

// ---------------------------------------------------------------------------
// Kernel 3: src/dst projections, pre-scaled into log2 domain.
//   srcT  = log2e * src
//   dstT  = log2e * dst
//   dstT5 = slope * log2e * dst
__global__ void gat_srcdst(const _Float16* __restrict__ ht, const float* __restrict__ attn_w,
                           float* __restrict__ srcT, float* __restrict__ dstT,
                           float* __restrict__ dstT5) {
    int n = blockIdx.x * 256 + threadIdx.x;  // 4096 nodes
    #pragma unroll
    for (int h = 0; h < GAT_HEADS; ++h) {
        float s = 0.f, d = 0.f;
        #pragma unroll
        for (int f = 0; f < GAT_HID; ++f) {
            float v = (float)ht[(h * GAT_HID + f) * GAT_N + n];
            s += v * attn_w[f];
            d += v * attn_w[GAT_HID + f];
        }
        s *= GAT_LOG2E;
        d *= GAT_LOG2E;
        srcT [h * GAT_N + n] = s;
        dstT [h * GAT_N + n] = d;
        dstT5[h * GAT_N + n] = GAT_SLOPE * d;
    }
}

// Kernel 4: dmax[h] = max_n dstT[h][n] (log2 domain; lrelu is monotone so the
// softmax row-max is exact: m_i = lrelu(s_i + dmax)).
__global__ void gat_dstmax(const float* __restrict__ dstT, float* __restrict__ dmax) {
    __shared__ float red[256];
    const int h = blockIdx.x;
    float m = -INFINITY;
    for (int n = threadIdx.x; n < GAT_N; n += 256)
        m = fmaxf(m, dstT[h * GAT_N + n]);
    red[threadIdx.x] = m;
    __syncthreads();
    for (int s = 128; s > 0; s >>= 1) {
        if (threadIdx.x < s) red[threadIdx.x] = fmaxf(red[threadIdx.x], red[threadIdx.x + s]);
        __syncthreads();
    }
    if (threadIdx.x == 0) dmax[h] = red[0];
}

// ---------------------------------------------------------------------------
// Kernel 5: fused flash-style attention, split-K over j (4 chunks of 1024).
// Block = 128 threads = 4 waves; grid = (itile 0..255) x (chunk 0..3); wave = head.
// Inner body per attention weight: add, add, max, v_exp_f32 -> f16 A-operand of
// v_wmma_f32_16x16x32_f16. Partials (C tile, Z row-sums) combine linearly since
// m_i is the exact global row max.
__global__ void gat_attn(const _Float16* __restrict__ ht, const float* __restrict__ srcT,
                         const float* __restrict__ dstT, const float* __restrict__ dstT5,
                         const float* __restrict__ dmax,
                         float* __restrict__ Cpart, float* __restrict__ Zpart) {
    const int wave  = threadIdx.x >> 5;      // head
    const int lane  = threadIdx.x & 31;
    const int h     = wave;
    const int itile = blockIdx.x & 255;
    const int chunk = blockIdx.x >> 8;       // 0..3
    const int i0    = itile * 16;
    const int r     = lane & 15;
    const bool hi   = lane >= 16;
    const int jbeg  = chunk * GAT_JLEN;

    const float s  = srcT[h * GAT_N + i0 + r];       // log2-domain src
    const float t0 = s + dmax[h];
    const float mi = fmaxf(t0, GAT_SLOPE * t0);      // exact row max (log2 domain)
    const float s1 = s - mi;                         // per-lane constants:
    const float s2 = GAT_SLOPE * s - mi;             // lrelu(s+d)-mi = max(s1+d, s2+0.2d)

    const int base0 = hi ? 16 : 0;                   // A K-chunk base
    const int col   = h * GAT_HID + r;               // B/C column (global feature)
    const _Float16* bptr = ht + col * GAT_N + jbeg + (hi ? 16 : 0);
    const float*    dp   = dstT  + h * GAT_N + jbeg + base0;
    const float*    dp5  = dstT5 + h * GAT_N + jbeg + base0;

    v8f acc = {};
    float z0 = 0.f, z1 = 0.f;
    #pragma unroll 2
    for (int j0 = 0; j0 < GAT_JLEN; j0 += 32) {
        v16h a;
        #pragma unroll
        for (int e = 0; e < 8; ++e) {
            float p = __builtin_amdgcn_exp2f(fmaxf(s1 + dp[j0 + e],
                                                   s2 + dp5[j0 + e]));
            z0 += p;
            a[e] = (_Float16)p;
            float q = __builtin_amdgcn_exp2f(fmaxf(s1 + dp[j0 + 8 + e],
                                                   s2 + dp5[j0 + 8 + e]));
            z1 += q;
            a[8 + e] = (_Float16)q;
        }
        v16h b = *(const v16h*)(bptr + j0);
        acc = __builtin_amdgcn_wmma_f32_16x16x32_f16(false, a, false, b,
                                                     (short)0, acc, false, false);
    }
    // Row sums: lane l and l+16 each hold half of row (l&15)'s p-sum.
    float z = z0 + z1;
    z += __shfl_xor(z, 16, 32);
    if (lane < 16)
        Zpart[(chunk * GAT_HEADS + h) * GAT_N + i0 + r] = z;

    const int rbase = hi ? 8 : 0;
    #pragma unroll
    for (int rr = 0; rr < 8; ++rr) {
        int row = i0 + rbase + rr;
        Cpart[(chunk * GAT_N + row) * GAT_FOUT + col] = acc[rr];
    }
}

// Kernel 6: combine split-K partials: out = sum_k C_k / sum_k Z_k.
__global__ void gat_reduce(const float* __restrict__ Cpart, const float* __restrict__ Zpart,
                           float* __restrict__ out) {
    int idx = blockIdx.x * 256 + threadIdx.x;        // 4096*64
    int row = idx >> 6, col = idx & 63;
    int h   = col >> 4;
    float c = 0.f, z = 0.f;
    #pragma unroll
    for (int k = 0; k < GAT_KCHUNK; ++k) {
        c += Cpart[(k * GAT_N + row) * GAT_FOUT + col];
        z += Zpart[(k * GAT_HEADS + h) * GAT_N + row];
    }
    out[idx] = c / z;
}

// ---------------------------------------------------------------------------
extern "C" void kernel_launch(void* const* d_in, const int* in_sizes, int n_in,
                              void* d_out, int out_size, void* d_ws, size_t ws_size,
                              hipStream_t stream) {
    const float* x      = (const float*)d_in[0];   // 4096*128
    // d_in[1] = adj : UNUSED by the reference (dense attention) -> never read
    const float* W      = (const float*)d_in[2];   // 128*64
    const float* attn_w = (const float*)d_in[3];   // 32
    float* out = (float*)d_out;                    // 4096*64

    _Float16* xh    = (_Float16*)d_ws;                     // 4096*128       (1 MB)
    _Float16* wt    = xh + GAT_N * GAT_FIN;                // 64*128         (16 KB)
    _Float16* ht    = wt + GAT_FOUT * GAT_FIN;             // 64*4096        (512 KB)
    float*    srcT  = (float*)(ht + GAT_FOUT * GAT_N);     // 4*4096
    float*    dstT  = srcT + GAT_HEADS * GAT_N;            // 4*4096
    float*    dstT5 = dstT + GAT_HEADS * GAT_N;            // 4*4096
    float*    dmax  = dstT5 + GAT_HEADS * GAT_N;           // 4
    float*    Zpart = dmax + 16;                           // 4*4*4096       (256 KB)
    float*    Cpart = Zpart + GAT_KCHUNK * GAT_HEADS * GAT_N; // 4*4096*64   (4 MB)

    gat_conv_x <<<(GAT_N * GAT_FIN) / 256, 256, 0, stream>>>(x, xh);
    gat_conv_w <<<(GAT_FOUT * GAT_FIN) / 256, 256, 0, stream>>>(W, wt);
    gat_proj   <<<GAT_N / 16, 128, 0, stream>>>(xh, wt, ht);
    gat_srcdst <<<GAT_N / 256, 256, 0, stream>>>(ht, attn_w, srcT, dstT, dstT5);
    gat_dstmax <<<GAT_HEADS, 256, 0, stream>>>(dstT, dmax);
    gat_attn   <<<(GAT_N / 16) * GAT_KCHUNK, 128, 0, stream>>>(ht, srcT, dstT, dstT5,
                                                               dmax, Cpart, Zpart);
    gat_reduce <<<(GAT_N * GAT_FOUT) / 256, 256, 0, stream>>>(Cpart, Zpart, out);
}
